// TransformerEncoder_47734266528278
// MI455X (gfx1250) — matrix operand order repaired and checked by reading, not compiled, not run on hardware
//
#include <hip/hip_runtime.h>
#include <hip/hip_bf16.h>
#include <stdint.h>

#define NB   32
#define TT   512
#define HH   512
#define NHD  8
#define DH   64
#define NT   (NB*TT)                 // 16384 rows
#define NTH  ((size_t)NT*HH)         // 8388608 elems
#define EPS  1e-5f
#define NEGC (-4294967295.0f)        // -(2^32)+1

typedef __bf16 bf16;
typedef bf16  v16bf __attribute__((ext_vector_type(16)));
typedef bf16  v8bf  __attribute__((ext_vector_type(8)));
typedef float v8f   __attribute__((ext_vector_type(8)));
typedef unsigned int u32x4 __attribute__((ext_vector_type(4)));

// ---------------- CDNA5 async global->LDS copy (ASYNCcnt path, §08) ----------------
// 16 bytes per lane, no VGPR round-trip. LDS address = low 32 bits of flat pointer.
__device__ __forceinline__ void async_copy_b128(bf16* lds, const bf16* gptr){
  unsigned laddr = (unsigned)(unsigned long long)lds;
  asm volatile("global_load_async_to_lds_b128 %0, %1, off"
               :: "v"(laddr), "v"(gptr) : "memory");
}
__device__ __forceinline__ void wait_async0(){
  asm volatile("s_wait_asynccnt 0x0" ::: "memory");
}

// ---------------- WMMA fragment loaders (CDNA5 §7.12.2 layouts) ----------------
// A 16x32 bf16: lane L<16 holds row L, K = k0+{0..7} (elems0-7), k0+{16..23} (elems8-15)
//               lane L>=16 same row, K = k0+{8..15}, k0+{24..31}
__device__ __forceinline__ v16bf frag_a(const bf16* base, int ld, int row0, int k0){
  int lane = threadIdx.x & 31;
  const bf16* p = base + (row0 + (lane & 15))*ld + k0 + ((lane >> 4) << 3);
  v16bf v;
#pragma unroll
  for (int j=0;j<8;++j){ v[j]=p[j]; v[j+8]=p[j+16]; }
  return v;
}
// B 32x16 bf16 (K x N): lane n<16 holds col n, K = k0+{0..7},{16..23}; lane n+16 K = k0+{8..15},{24..31}
__device__ __forceinline__ v16bf frag_b(const bf16* base, int ld, int col0, int k0){
  int lane = threadIdx.x & 31;
  const bf16* p = base + (col0 + (lane & 15))*ld + k0 + ((lane >> 4) << 3);
  v16bf v;
#pragma unroll
  for (int j=0;j<8;++j){ v[j]=p[j]; v[j+8]=p[j+16]; }
  return v;
}

__device__ __forceinline__ float red16_max(float v){
#pragma unroll
  for (int m=1;m<16;m<<=1) v = fmaxf(v, __shfl_xor(v,m,32));
  return v;
}
__device__ __forceinline__ float red16_sum(float v){
#pragma unroll
  for (int m=1;m<16;m<<=1) v += __shfl_xor(v,m,32);
  return v;
}
__device__ __forceinline__ float red32_sum(float v){
#pragma unroll
  for (int m=1;m<32;m<<=1) v += __shfl_xor(v,m,32);
  return v;
}

// ---------------- weight f32 -> bf16 ----------------
__global__ __launch_bounds__(256) void cvt_bf16(const float* __restrict__ a,
                                                bf16* __restrict__ o, int n){
  int i = blockIdx.x*256 + threadIdx.x;
  if (i < n) o[i] = (bf16)a[i];
}

// ---------------- prep: seq*mask, LN1, masks ----------------
__global__ __launch_bounds__(256) void prep_kernel(
    const float* __restrict__ seq, const float* __restrict__ mask,
    const float* __restrict__ g1, const float* __restrict__ b1,
    float* __restrict__ seqm, bf16* __restrict__ seqm_b,
    float* __restrict__ qf,  bf16* __restrict__ q_b,
    float* __restrict__ km,  float* __restrict__ qm){
  int row  = blockIdx.x*8 + (threadIdx.x >> 5);
  int lane = threadIdx.x & 31;
  const float* src = seq + (size_t)row*HH;
  float mk = mask[row];
  float x[16]; float s = 0.f;
#pragma unroll
  for (int j=0;j<16;++j){ x[j] = src[lane + 32*j] * mk; s += x[j]; }
  s = red32_sum(s);
  if (lane == 0) km[row] = (s != 0.f) ? 1.f : 0.f;     // sign(|sum(keys)|)
  float mean = s * (1.f/HH);
  float vs = 0.f;
#pragma unroll
  for (int j=0;j<16;++j){ float d = x[j]-mean; vs += d*d; }
  vs = red32_sum(vs) * (1.f/HH);
  float rstd = rsqrtf(vs + EPS);
  float qs = 0.f;
#pragma unroll
  for (int j=0;j<16;++j){
    int c = lane + 32*j;
    size_t i = (size_t)row*HH + c;
    float q = (x[j]-mean)*rstd*g1[c] + b1[c];
    seqm[i] = x[j];  seqm_b[i] = (bf16)x[j];
    qf[i]   = q;     q_b[i]    = (bf16)q;
    qs += q;
  }
  qs = red32_sum(qs);
  if (lane == 0) qm[row] = (qs != 0.f) ? 1.f : 0.f;    // sign(|sum(queries)|)
}

// ---------------- tiled WMMA GEMM: C[m,n] = act(sum_k A[m,k]*W[n,k] + bias[n]) ----------------
// block 256 = 8 waves, tile 128(M) x 64(N), K-step 64, double-buffered async-LDS pipeline
#define ATILE (128*72)
#define BTILE (64*72)
__global__ __launch_bounds__(256) void gemm_bf16_wmma(
    const bf16* __restrict__ A, const bf16* __restrict__ W,
    const float* __restrict__ bias,
    float* __restrict__ outF, bf16* __restrict__ outB,
    float bscale, int relu){
  __shared__ bf16 As[2*ATILE];
  __shared__ bf16 Bs[2*BTILE];
  int m0 = blockIdx.y*128, n0 = blockIdx.x*64;
  int tid = threadIdx.x, w = tid>>5, lane = tid&31;
  int lo = lane&15, hi = lane>>4;
  v8f acc[4] = {};

  // prologue: async-load k0=0 tiles into buffer 0
  for (int i=tid;i<1024;i+=256){
    int r=i>>3, c=(i&7)*8;
    async_copy_b128(As + r*72 + c, A + (size_t)(m0+r)*HH + c);
  }
  for (int i=tid;i<512;i+=256){
    int r=i>>3, c=(i&7)*8;
    async_copy_b128(Bs + r*72 + c, W + (size_t)(n0+r)*HH + c);
  }
  wait_async0();
  __syncthreads();

  int cur = 0;
  for (int k0=0;k0<HH;k0+=64){
    const bf16* as = As + cur*ATILE;
    const bf16* bs = Bs + cur*BTILE;
    // issue next k-tile into the other buffer while computing this one
    if (k0 + 64 < HH){
      bf16* an = As + (cur^1)*ATILE;
      bf16* bn = Bs + (cur^1)*BTILE;
      int kn = k0 + 64;
      for (int i=tid;i<1024;i+=256){
        int r=i>>3, c=(i&7)*8;
        async_copy_b128(an + r*72 + c, A + (size_t)(m0+r)*HH + kn + c);
      }
      for (int i=tid;i<512;i+=256){
        int r=i>>3, c=(i&7)*8;
        async_copy_b128(bn + r*72 + c, W + (size_t)(n0+r)*HH + kn + c);
      }
    }
    // batch fragment loads, then back-to-back WMMAs
    v16bf a0 = frag_a(as, 72, 16*w, 0);
    v16bf a1 = frag_a(as, 72, 16*w, 32);
    v16bf bb[8];
#pragma unroll
    for (int g=0; g<4; ++g){
      bb[g]   = frag_b(bs, 72, g*16, 0);
      bb[4+g] = frag_b(bs, 72, g*16, 32);
    }
#pragma unroll
    for (int g=0; g<4; ++g)
      acc[g] = __builtin_amdgcn_wmma_f32_16x16x32_bf16(
          false, a0, false, bb[g], (short)0, acc[g], false, false);
#pragma unroll
    for (int g=0; g<4; ++g)
      acc[g] = __builtin_amdgcn_wmma_f32_16x16x32_bf16(
          false, a1, false, bb[4+g], (short)0, acc[g], false, false);
    if (k0 + 64 < HH) wait_async0();
    __syncthreads();
    cur ^= 1;
  }
#pragma unroll
  for (int g=0; g<4; ++g){
    int col = n0 + g*16 + lo;
    float bv = bias[col];
#pragma unroll
    for (int r=0;r<8;++r){
      int row = m0 + 16*w + 8*hi + r;
      float v = acc[g][r] + bv;
      if (relu) v = fmaxf(v, 0.f);
      size_t i = (size_t)row*HH + col;
      if (outF) outF[i] = v;
      if (outB) outB[i] = (bf16)(v*bscale);
    }
  }
}

// ---------------- flash attention (causal + key/query mask), per (n,h,64-q-tile) ----------------
__global__ __launch_bounds__(128) void attn_flash(
    const bf16* __restrict__ Q, const bf16* __restrict__ K, const bf16* __restrict__ V,
    const float* __restrict__ km, const float* __restrict__ qm,
    float* __restrict__ outp){
  __shared__ bf16 Qs[64*72];
  __shared__ bf16 Ks[64*72];
  __shared__ bf16 Vt[64*72];          // transposed: [d][k]
  __shared__ bf16 Pb[4*16*72];        // per-wave C->A bounce buffer
  int qt = blockIdx.x, h = blockIdx.y, n = blockIdx.z;
  int tid=threadIdx.x, w=tid>>5, lane=tid&31, lo=lane&15, hi=lane>>4;
  int q0 = qt*64;
  size_t base = ((size_t)n*TT)*HH + h*DH;

  // Q tile via async-LDS
  for (int i=tid;i<512;i+=128){
    int r=i>>3, c=(i&7)*8;
    async_copy_b128(Qs + r*72 + c, Q + base + (size_t)(q0+r)*HH + c);
  }
  wait_async0();
  __syncthreads();
  v16bf aq0 = frag_a(Qs, 72, 16*w, 0);
  v16bf aq1 = frag_a(Qs, 72, 16*w, 32);

  v8f acc[4] = {};
  float ms[8], ls[8];
#pragma unroll
  for (int r=0;r<8;++r){ ms[r] = -3.0e38f; ls[r] = 0.f; }

  for (int kt=0; kt<=qt; ++kt){
    int k0g = kt*64;
    __syncthreads();                       // previous tile's LDS fully consumed
    // K tile async, V tile transposed (vector global load + strided ds stores)
    for (int i=tid;i<512;i+=128){
      int r=i>>3, c=(i&7)*8;
      async_copy_b128(Ks + r*72 + c, K + base + (size_t)(k0g+r)*HH + c);
    }
    for (int i=tid;i<512;i+=128){
      int kk=i>>3, dd=(i&7)*8;
      v8bf vv = *(const v8bf*)(V + base + (size_t)(k0g+kk)*HH + dd);
#pragma unroll
      for (int j=0;j<8;++j) Vt[(dd+j)*72 + kk] = vv[j];
    }
    wait_async0();
    __syncthreads();

    // S = (Q*scale) @ K^T   (scale folded into Q's bf16 store)
    v16bf kb[8];
#pragma unroll
    for (int g=0; g<4; ++g){
      kb[g]   = frag_b(Ks, 72, g*16, 0);
      kb[4+g] = frag_b(Ks, 72, g*16, 32);
    }
    v8f s[4];
#pragma unroll
    for (int g=0; g<4; ++g){
      v8f z = {};
      z = __builtin_amdgcn_wmma_f32_16x16x32_bf16(false, aq0, false, kb[g],   (short)0, z, false, false);
      s[g] = z;
    }
#pragma unroll
    for (int g=0; g<4; ++g)
      s[g] = __builtin_amdgcn_wmma_f32_16x16x32_bf16(false, aq1, false, kb[4+g], (short)0, s[g], false, false);

    float kmv[4]; int colg[4];
#pragma unroll
    for (int g=0; g<4; ++g){
      colg[g] = k0g + g*16 + lo;
      kmv[g]  = km[(size_t)n*TT + colg[g]];
    }
    // online softmax update
#pragma unroll
    for (int r=0;r<8;++r){
      int rowg = q0 + 16*w + 8*hi + r;
      float rm = -3.0e38f;
#pragma unroll
      for (int g=0; g<4; ++g){
        float v = s[g][r];
        v = (kmv[g]==0.f || colg[g] > rowg) ? NEGC : v;
        s[g][r] = v;
        rm = fmaxf(rm, v);
      }
      rm = red16_max(rm);
      float mnew  = fmaxf(ms[r], rm);
      float alpha = __expf(ms[r] - mnew);
      ms[r] = mnew;
      float rs = 0.f;
#pragma unroll
      for (int g=0; g<4; ++g){
        float p = __expf(s[g][r] - mnew);
        Pb[(w*16 + 8*hi + r)*72 + g*16 + lo] = (bf16)p;
        rs += p;
      }
      rs = red16_sum(rs);
      ls[r] = ls[r]*alpha + rs;
#pragma unroll
      for (int g=0; g<4; ++g) acc[g][r] *= alpha;
    }
    // wave-private LDS bounce: C layout -> A layout; explicit CDNA5 DS wait
    asm volatile("s_wait_dscnt 0" ::: "memory");
    v16bf ap0 = frag_a(Pb + w*16*72, 72, 0, 0);
    v16bf ap1 = frag_a(Pb + w*16*72, 72, 0, 32);
    v16bf vb[8];
#pragma unroll
    for (int g=0; g<4; ++g){
      vb[g]   = frag_b(Vt, 72, g*16, 0);
      vb[4+g] = frag_b(Vt, 72, g*16, 32);
    }
#pragma unroll
    for (int g=0; g<4; ++g)
      acc[g] = __builtin_amdgcn_wmma_f32_16x16x32_bf16(false, ap0, false, vb[g],   (short)0, acc[g], false, false);
#pragma unroll
    for (int g=0; g<4; ++g)
      acc[g] = __builtin_amdgcn_wmma_f32_16x16x32_bf16(false, ap1, false, vb[4+g], (short)0, acc[g], false, false);
  }
  // epilogue: 1/l and query-mask
#pragma unroll
  for (int r=0;r<8;++r){
    int rowg = q0 + 16*w + 8*hi + r;
    float f = qm[(size_t)n*TT + rowg] / ls[r];
#pragma unroll
    for (int g=0; g<4; ++g)
      outp[base + (size_t)rowg*HH + g*16 + lo] = acc[g][r]*f;
  }
}

// ---------------- seq2 = seqm + attn + queries; x = LN2(seq2) ----------------
__global__ __launch_bounds__(256) void resid_ln2_kernel(
    const float* __restrict__ seqm, const float* __restrict__ attn,
    const float* __restrict__ qf,
    const float* __restrict__ g2, const float* __restrict__ b2,
    float* __restrict__ seq2, float* __restrict__ xf, bf16* __restrict__ xb){
  int row  = blockIdx.x*8 + (threadIdx.x >> 5);
  int lane = threadIdx.x & 31;
  float t[16]; float s = 0.f;
#pragma unroll
  for (int j=0;j<16;++j){
    size_t i = (size_t)row*HH + lane + 32*j;
    t[j] = seqm[i] + attn[i] + qf[i];
    seq2[i] = t[j];
    s += t[j];
  }
  s = red32_sum(s);
  float mean = s*(1.f/HH), vs = 0.f;
#pragma unroll
  for (int j=0;j<16;++j){ float d=t[j]-mean; vs += d*d; }
  vs = red32_sum(vs)*(1.f/HH);
  float rstd = rsqrtf(vs + EPS);
#pragma unroll
  for (int j=0;j<16;++j){
    int c = lane + 32*j;
    size_t i = (size_t)row*HH + c;
    float x = (t[j]-mean)*rstd*g2[c] + b2[c];
    xf[i] = x; xb[i] = (bf16)x;
  }
}

// ---------------- ff = LNf(h2+x)*mask; out = LN3(seq2+ff) ----------------
__global__ __launch_bounds__(256) void final_kernel(
    const float* __restrict__ h2, const float* __restrict__ xf,
    const float* __restrict__ seq2, const float* __restrict__ mask,
    const float* __restrict__ gf, const float* __restrict__ bf,
    const float* __restrict__ g3, const float* __restrict__ b3,
    float* __restrict__ out){
  int row  = blockIdx.x*8 + (threadIdx.x >> 5);
  int lane = threadIdx.x & 31;
  float mk = mask[row];
  float t[16]; float s = 0.f;
#pragma unroll
  for (int j=0;j<16;++j){
    size_t i = (size_t)row*HH + lane + 32*j;
    t[j] = h2[i] + xf[i];
    s += t[j];
  }
  s = red32_sum(s);
  float mean = s*(1.f/HH), vs = 0.f;
#pragma unroll
  for (int j=0;j<16;++j){ float d=t[j]-mean; vs += d*d; }
  vs = red32_sum(vs)*(1.f/HH);
  float rstd = rsqrtf(vs + EPS);
  float u[16]; s = 0.f;
#pragma unroll
  for (int j=0;j<16;++j){
    int c = lane + 32*j;
    size_t i = (size_t)row*HH + c;
    float ff = ((t[j]-mean)*rstd*gf[c] + bf[c]) * mk;
    u[j] = seq2[i] + ff;
    s += u[j];
  }
  s = red32_sum(s);
  mean = s*(1.f/HH); vs = 0.f;
#pragma unroll
  for (int j=0;j<16;++j){ float d=u[j]-mean; vs += d*d; }
  vs = red32_sum(vs)*(1.f/HH);
  rstd = rsqrtf(vs + EPS);
#pragma unroll
  for (int j=0;j<16;++j){
    int c = lane + 32*j;
    out[(size_t)row*HH + c] = (u[j]-mean)*rstd*g3[c] + b3[c];
  }
}

extern "C" void kernel_launch(void* const* d_in, const int* in_sizes, int n_in,
                              void* d_out, int out_size, void* d_ws, size_t ws_size,
                              hipStream_t stream) {
  (void)in_sizes; (void)n_in; (void)out_size; (void)ws_size;
  const float* seq  = (const float*)d_in[0];
  const float* mask = (const float*)d_in[1];
  const float* Wq = (const float*)d_in[2];  const float* bq = (const float*)d_in[3];
  const float* Wk = (const float*)d_in[4];  const float* bk = (const float*)d_in[5];
  const float* Wv = (const float*)d_in[6];  const float* bv = (const float*)d_in[7];
  const float* W1 = (const float*)d_in[8];  const float* b1 = (const float*)d_in[9];
  const float* W2 = (const float*)d_in[10]; const float* b2 = (const float*)d_in[11];
  const float* ln1g=(const float*)d_in[12]; const float* ln1b=(const float*)d_in[13];
  const float* ln2g=(const float*)d_in[14]; const float* ln2b=(const float*)d_in[15];
  const float* ln3g=(const float*)d_in[16]; const float* ln3b=(const float*)d_in[17];
  const float* lnfg=(const float*)d_in[18]; const float* lnfb=(const float*)d_in[19];
  float* out = (float*)d_out;

  char* p = (char*)d_ws;
  float* seqm = (float*)p; p += NTH*4;
  float* qf   = (float*)p; p += NTH*4;
  float* attn = (float*)p; p += NTH*4;
  float* seq2 = (float*)p; p += NTH*4;
  float* xf   = (float*)p; p += NTH*4;
  float* h2   = (float*)p; p += NTH*4;
  float* km   = (float*)p; p += (size_t)NT*4;
  float* qm   = (float*)p; p += (size_t)NT*4;
  bf16* seqm_b=(bf16*)p; p += NTH*2;
  bf16* q_b  = (bf16*)p; p += NTH*2;
  bf16* Qb   = (bf16*)p; p += NTH*2;
  bf16* Kb   = (bf16*)p; p += NTH*2;
  bf16* Vb   = (bf16*)p; p += NTH*2;
  bf16* xb   = (bf16*)p; p += NTH*2;
  bf16* h1b  = (bf16*)p; p += NTH*2;
  bf16* wq_b = (bf16*)p; p += (size_t)HH*HH*2;
  bf16* wk_b = (bf16*)p; p += (size_t)HH*HH*2;
  bf16* wv_b = (bf16*)p; p += (size_t)HH*HH*2;
  bf16* w1_b = (bf16*)p; p += (size_t)HH*HH*2;
  bf16* w2_b = (bf16*)p; p += (size_t)HH*HH*2;

  const int WN = HH*HH;
  cvt_bf16<<<(WN+255)/256, 256, 0, stream>>>(Wq, wq_b, WN);
  cvt_bf16<<<(WN+255)/256, 256, 0, stream>>>(Wk, wk_b, WN);
  cvt_bf16<<<(WN+255)/256, 256, 0, stream>>>(Wv, wv_b, WN);
  cvt_bf16<<<(WN+255)/256, 256, 0, stream>>>(W1, w1_b, WN);
  cvt_bf16<<<(WN+255)/256, 256, 0, stream>>>(W2, w2_b, WN);

  prep_kernel<<<NT/8, 256, 0, stream>>>(seq, mask, ln1g, ln1b,
                                        seqm, seqm_b, qf, q_b, km, qm);

  dim3 gg(HH/64, NT/128);
  const float qscale = 0.04419417382415922f;   // 1/sqrt(512)
  gemm_bf16_wmma<<<gg, 256, 0, stream>>>(q_b,    wq_b, bq, nullptr, Qb,  qscale, 0);
  gemm_bf16_wmma<<<gg, 256, 0, stream>>>(seqm_b, wk_b, bk, nullptr, Kb,  1.0f,   0);
  gemm_bf16_wmma<<<gg, 256, 0, stream>>>(seqm_b, wv_b, bv, nullptr, Vb,  1.0f,   0);

  dim3 ga(TT/64, NHD, NB);
  attn_flash<<<ga, 128, 0, stream>>>(Qb, Kb, Vb, km, qm, attn);

  resid_ln2_kernel<<<NT/8, 256, 0, stream>>>(seqm, attn, qf, ln2g, ln2b,
                                             seq2, xf, xb);

  gemm_bf16_wmma<<<gg, 256, 0, stream>>>(xb,  w1_b, b1, nullptr, h1b, 1.0f, 1);
  gemm_bf16_wmma<<<gg, 256, 0, stream>>>(h1b, w2_b, b2, h2, nullptr, 1.0f, 0);

  final_kernel<<<NT/8, 256, 0, stream>>>(h2, xf, seq2, mask,
                                         lnfg, lnfb, ln3g, ln3b, out);
}
